// BinaryDense_44908178047611
// MI455X (gfx1250) — 8-bit matrix operand order repaired and checked by reading, compiled, not run on hardware
//
#include <hip/hip_runtime.h>

typedef __attribute__((ext_vector_type(8))) int v8i;

#define B_DIM   8192
#define D_IN    2048
#define UNITS   2048

#define BM 128
#define BN 128
#define BK 64

union FragU {
    unsigned long long d[4];
    v8i v;
};

__global__ __launch_bounds__(256) void binarize_rows(const float* __restrict__ in,
                                                     char* __restrict__ out,
                                                     int n4) {
    int i = blockIdx.x * blockDim.x + threadIdx.x;
    if (i >= n4) return;
    float4 v = ((const float4*)in)[i];
    union { char c[4]; int u; } p;
    p.c[0] = (v.x >= 0.0f) ? 1 : -1;
    p.c[1] = (v.y >= 0.0f) ? 1 : -1;
    p.c[2] = (v.z >= 0.0f) ? 1 : -1;
    p.c[3] = (v.w >= 0.0f) ? 1 : -1;
    ((int*)out)[i] = p.u;
}

__global__ __launch_bounds__(256) void binarize_transpose(const float* __restrict__ w,
                                                          char* __restrict__ wt) {
    __shared__ char tile[32][33];
    int n0 = blockIdx.x * 32;
    int k0 = blockIdx.y * 32;
    #pragma unroll
    for (int j = threadIdx.y; j < 32; j += 8) {
        float v = w[(size_t)(k0 + j) * UNITS + n0 + threadIdx.x];
        tile[j][threadIdx.x] = (v >= 0.0f) ? 1 : -1;
    }
    __syncthreads();
    #pragma unroll
    for (int j = threadIdx.y; j < 32; j += 8) {
        wt[(size_t)(n0 + j) * D_IN + k0 + threadIdx.x] = tile[threadIdx.x][j];
    }
}

__global__ __launch_bounds__(256) void bgemm_iu8(const char* __restrict__ xb,
                                                 const char* __restrict__ wtb,
                                                 float* __restrict__ out) {
    alignas(16) __shared__ char sA[BM * BK];
    alignas(16) __shared__ char sB[BN * BK];

    const int tid  = threadIdx.x;
    const int wave = tid >> 5;
    const int lane = tid & 31;
    const int mw   = wave & 3;
    const int nw   = wave >> 2;
    const int r    = lane & 15;
    const int h    = lane >> 4;

    const int row0 = blockIdx.y * BM;
    const int col0 = blockIdx.x * BN;

    v8i acc[2][4];
    #pragma unroll
    for (int tm = 0; tm < 2; ++tm)
        #pragma unroll
        for (int tn = 0; tn < 4; ++tn)
            acc[tm][tn] = (v8i)(0);

    const int srow = tid >> 2;
    const int soff = (tid & 3) << 4;

    for (int k0 = 0; k0 < D_IN; k0 += BK) {
        #pragma unroll
        for (int it = 0; it < 2; ++it) {
            int rr = srow + it * 64;
            *(int4*)&sA[rr * BK + soff] =
                *(const int4*)&xb[(size_t)(row0 + rr) * D_IN + k0 + soff];
            *(int4*)&sB[rr * BK + soff] =
                *(const int4*)&wtb[(size_t)(col0 + rr) * D_IN + k0 + soff];
        }
        if (k0 + BK < D_IN) {
            __builtin_prefetch(&xb[(size_t)(row0 + srow) * D_IN + k0 + BK + soff], 0, 3);
        }
        __syncthreads();

        v8i a[2];
        #pragma unroll
        for (int tm = 0; tm < 2; ++tm) {
            const char* base = &sA[(mw * 32 + tm * 16 + r) * BK];
            FragU fa;
            #pragma unroll
            for (int j = 0; j < 4; ++j)
                fa.d[j] = *(const unsigned long long*)(base + (j >> 1) * 32 + (j & 1) * 8 + h * 16);
            a[tm] = fa.v;
        }
        v8i b[4];
        #pragma unroll
        for (int tn = 0; tn < 4; ++tn) {
            const char* base = &sB[(nw * 64 + tn * 16 + r) * BK + h * 16];
            FragU fb;
            fb.d[0] = *(const unsigned long long*)(base);
            fb.d[1] = *(const unsigned long long*)(base + 8);
            fb.d[2] = *(const unsigned long long*)(base + 32);
            fb.d[3] = *(const unsigned long long*)(base + 40);
            b[tn] = fb.v;
        }

        #pragma unroll
        for (int tm = 0; tm < 2; ++tm)
            #pragma unroll
            for (int tn = 0; tn < 4; ++tn)
                acc[tm][tn] = __builtin_amdgcn_wmma_i32_16x16x64_iu8(
                    true, a[tm], true, b[tn], acc[tm][tn], false, false);

        __syncthreads();
    }

    #pragma unroll
    for (int tm = 0; tm < 2; ++tm) {
        #pragma unroll
        for (int tn = 0; tn < 4; ++tn) {
            int coln = col0 + nw * 64 + tn * 16 + r;
            #pragma unroll
            for (int g = 0; g < 8; ++g) {
                int rowm = row0 + mw * 32 + tm * 16 + g + h * 8;
                out[(size_t)rowm * UNITS + coln] = (float)acc[tm][tn][g];
            }
        }
    }
}

extern "C" void kernel_launch(void* const* d_in, const int* in_sizes, int n_in,
                              void* d_out, int out_size, void* d_ws, size_t ws_size,
                              hipStream_t stream) {
    const float* x = (const float*)d_in[0];
    const float* w = (const float*)d_in[1];
    float* out = (float*)d_out;

    char* xb  = (char*)d_ws;
    char* wtb = xb + (size_t)B_DIM * D_IN;

    {
        int n4 = (B_DIM * D_IN) / 4;
        binarize_rows<<<(n4 + 255) / 256, 256, 0, stream>>>(x, xb, n4);
    }
    {
        dim3 tb(32, 8);
        dim3 grid(UNITS / 32, D_IN / 32);
        binarize_transpose<<<grid, tb, 0, stream>>>(w, wtb);
    }
    {
        dim3 grid(UNITS / BN, B_DIM / BM);
        bgemm_iu8<<<grid, 256, 0, stream>>>(xb, wtb, out);
    }
}
